// PointDecoder_70746701300259
// MI455X (gfx1250) — matrix operand order repaired and checked by reading, not compiled, not run on hardware
//
#include <hip/hip_runtime.h>
#include <hip/hip_bf16.h>
#include <math.h>

#define BATCH 32
#define DD 256
#define HW 4096          // 64*64
#define THRV 0.1f

typedef _Float16 h8_t  __attribute__((ext_vector_type(8)));
typedef _Float16 h16_t __attribute__((ext_vector_type(16)));
typedef float    f8_t  __attribute__((ext_vector_type(8)));

// ---------------------------------------------------------------------------
// WMMA GEMM:  C[M,N] = A[M,K](f16) * W[K,N]  using WT[N,K](f16), f32 accum.
// Each wave computes a 16 x (16*NT) strip of C: one A fragment is reused
// across NT B fragments per k-step (A read exactly once from HBM).
// K is a compile-time constant so all A/B accesses are base + immediate
// offset (no per-tile pointer arithmetic, better load clustering).
// 4 waves per block -> block covers 64 x (16*NT). M%64==0, N%(16*NT)==0.
// ---------------------------------------------------------------------------
template <int NT, int K>
__global__ void gemm_wmma_f16(const _Float16* __restrict__ A,
                              const _Float16* __restrict__ WT,
                              const float* __restrict__ bias,
                              float* __restrict__ C,
                              int M, int N) {
    int wave = threadIdx.x >> 5;
    int lane = threadIdx.x & 31;
    int mt = blockIdx.y * 4 + wave;
    if (mt * 16 >= M) return;                     // wave-uniform
    int sub = lane >> 4;
    int rc  = lane & 15;
    const _Float16* Ar  = A  + (size_t)(mt * 16 + rc) * K + 16 * sub;
    const _Float16* Br0 = WT + (size_t)(blockIdx.x * NT * 16 + rc) * K + 16 * sub;
    f8_t acc[NT];
#pragma unroll
    for (int nt = 0; nt < NT; ++nt) acc[nt] = (f8_t){0.f,0.f,0.f,0.f,0.f,0.f,0.f,0.f};
#pragma unroll
    for (int k0 = 0; k0 < K; k0 += 32) {
        h8_t alo = *(const h8_t*)(Ar + k0);
        h8_t ahi = *(const h8_t*)(Ar + k0 + 8);
        h16_t a;
#pragma unroll
        for (int e = 0; e < 8; ++e) { a[e] = alo[e]; a[8 + e] = ahi[e]; }
#pragma unroll
        for (int nt = 0; nt < NT; ++nt) {
            h16_t b = *(const h16_t*)(Br0 + nt * 16 * K + k0);   // immediate offsets
            acc[nt] = __builtin_amdgcn_wmma_f32_16x16x32_f16(
                false, a, false, b, (short)0, acc[nt], false, false);
        }
    }
#pragma unroll
    for (int nt = 0; nt < NT; ++nt) {
        int col = blockIdx.x * NT * 16 + nt * 16 + rc;
        float bv = bias ? bias[col] : 0.f;
#pragma unroll
        for (int r = 0; r < 8; ++r) {
            int row = mt * 16 + r + 8 * sub;
            C[(size_t)row * N + col] = acc[nt][r] + bv;
        }
    }
}

// W[K,N] f32 -> WT[N,K] f16
__global__ void transpose_w(const float* __restrict__ W, _Float16* __restrict__ WT,
                            int K, int N) {
    int i = blockIdx.x * 256 + threadIdx.x;
    if (i >= K * N) return;
    int n = i % N, k = i / N;
    WT[(size_t)n * K + k] = (_Float16)W[(size_t)k * N + n];
}

// ---------------------------------------------------------------------------
// Positional encoding: pe[n, d] for n in 64x64, D=256
// ---------------------------------------------------------------------------
__global__ void pe_kernel(float* __restrict__ pe, const float* __restrict__ gauss) {
    int i = blockIdx.x * 256 + threadIdx.x;      // 4096*128
    if (i >= HW * 128) return;
    int n = i >> 7, d = i & 127;
    int y = n >> 6, x = n & 63;
    float cx = (x + 0.5f) / 64.f * 2.f - 1.f;
    float cy = (y + 0.5f) / 64.f * 2.f - 1.f;
    float pr = 6.28318530718f * (cx * gauss[d] + cy * gauss[128 + d]);
    pe[(size_t)n * DD + d]       = sinf(pr);
    pe[(size_t)n * DD + 128 + d] = cosf(pr);
}

// img (B,D,H*W) -> keys (B,HW,D) f32 + f16 + f16(keys+pe), tiled transpose
__global__ void keys_init(const float* __restrict__ img, const float* __restrict__ pe,
                          float* __restrict__ keys, _Float16* __restrict__ kh,
                          _Float16* __restrict__ kpeh) {
    __shared__ float tile[32][33];
    int b = blockIdx.z, d0 = blockIdx.y * 32, n0 = blockIdx.x * 32;
    int tx = threadIdx.x, ty = threadIdx.y;      // 32 x 8
#pragma unroll
    for (int i = 0; i < 4; ++i) {
        int d = d0 + ty + 8 * i;
        tile[ty + 8 * i][tx] = img[((size_t)b * DD + d) * HW + n0 + tx];
    }
    __syncthreads();
#pragma unroll
    for (int i = 0; i < 4; ++i) {
        int n = n0 + ty + 8 * i, d = d0 + tx;
        float v = tile[tx][ty + 8 * i];
        size_t o = ((size_t)b * HW + n) * DD + d;
        keys[o] = v;
        kh[o] = (_Float16)v;
        kpeh[o] = (_Float16)(v + pe[(size_t)n * DD + d]);
    }
}

__global__ void copy_queries(float* __restrict__ q, const float* __restrict__ tok) {
    int i = blockIdx.x * 256 + threadIdx.x;
    if (i < BATCH * DD) q[i] = tok[i & 255];
}

// out[b,:dout] = (in[b,:din] (+addvec)) @ W[din,dout] + bias  (opt relu)
__global__ void lin_small(const float* __restrict__ in, const float* __restrict__ W,
                          const float* __restrict__ bias, float* __restrict__ out,
                          int din, int dout, const float* __restrict__ addvec, int relu) {
    __shared__ float s[256];
    int b = blockIdx.x, t = threadIdx.x;
    if (t < din) s[t] = in[b * din + t] + (addvec ? addvec[t] : 0.f);
    __syncthreads();
    if (t < dout) {
        float acc = bias[t];
        for (int d = 0; d < din; ++d) acc += s[d] * W[d * dout + t];
        if (relu) acc = fmaxf(acc, 0.f);
        out[b * dout + t] = acc;
    }
}

// tgt = LN(replace ? add : tgt+add) over 256 dims
__global__ void res_ln(float* __restrict__ tgt, const float* __restrict__ add,
                       const float* __restrict__ sc, const float* __restrict__ bi,
                       int replace) {
    __shared__ float red[256];
    int b = blockIdx.x, t = threadIdx.x;
    float v = add[b * DD + t] + (replace ? 0.f : tgt[b * DD + t]);
    red[t] = v; __syncthreads();
    for (int st = 128; st > 0; st >>= 1) { if (t < st) red[t] += red[t + st]; __syncthreads(); }
    float m = red[0] / 256.f; __syncthreads();
    float d = v - m;
    red[t] = d * d; __syncthreads();
    for (int st = 128; st > 0; st >>= 1) { if (t < st) red[t] += red[t + st]; __syncthreads(); }
    float var = red[0] / 256.f;
    tgt[b * DD + t] = d * rsqrtf(var + 1e-5f) * sc[t] + bi[t];
}

// fused MLP (256->2048 relu ->256) + residual + LN
__global__ void mlp_kernel(float* __restrict__ q,
                           const float* __restrict__ w1, const float* __restrict__ b1,
                           const float* __restrict__ w2, const float* __restrict__ b2,
                           const float* __restrict__ sc, const float* __restrict__ bi) {
    __shared__ float qs[256];
    __shared__ float hid[2048];
    __shared__ float red[256];
    int b = blockIdx.x, t = threadIdx.x;
    qs[t] = q[b * DD + t];
    __syncthreads();
    for (int j = 0; j < 8; ++j) {
        int u = t + 256 * j;
        float a = b1[u];
        for (int d = 0; d < 256; ++d) a += qs[d] * w1[d * 2048 + u];
        hid[u] = fmaxf(a, 0.f);
    }
    __syncthreads();
    float o = b2[t];
    for (int u = 0; u < 2048; ++u) o += hid[u] * w2[u * 256 + t];
    float v = qs[t] + o;
    red[t] = v; __syncthreads();
    for (int st = 128; st > 0; st >>= 1) { if (t < st) red[t] += red[t + st]; __syncthreads(); }
    float m = red[0] / 256.f; __syncthreads();
    float d = v - m;
    red[t] = d * d; __syncthreads();
    for (int st = 128; st > 0; st >>= 1) { if (t < st) red[t] += red[t + st]; __syncthreads(); }
    float var = red[0] / 256.f;
    q[b * DD + t] = d * rsqrtf(var + 1e-5f) * sc[t] + bi[t];
}

// 1-query x 4096-keys attention, 8 heads of dh=16. block = (b,h).
__global__ void attn_t2i(const float* __restrict__ q32, const float* __restrict__ k,
                         const float* __restrict__ v, float* __restrict__ outp) {
    __shared__ float probs[4096];
    __shared__ float red[256];
    int bh = blockIdx.x; int b = bh >> 3, h = bh & 7; int t = threadIdx.x;
    float qv[16];
#pragma unroll
    for (int d = 0; d < 16; ++d) qv[d] = q32[b * 128 + h * 16 + d];
    float lmax = -1e30f;
    for (int n = t; n < HW; n += 256) {
        const float* kp = k + ((size_t)b * HW + n) * 128 + h * 16;
        float sccr = 0.f;
#pragma unroll
        for (int d = 0; d < 16; ++d) sccr += qv[d] * kp[d];
        sccr *= 0.25f;                             // 1/sqrt(16)
        probs[n] = sccr;
        lmax = fmaxf(lmax, sccr);
    }
    red[t] = lmax; __syncthreads();
    for (int st = 128; st > 0; st >>= 1) { if (t < st) red[t] = fmaxf(red[t], red[t + st]); __syncthreads(); }
    float m = red[0]; __syncthreads();
    float lsum = 0.f;
    for (int n = t; n < HW; n += 256) { float p = expf(probs[n] - m); probs[n] = p; lsum += p; }
    red[t] = lsum; __syncthreads();
    for (int st = 128; st > 0; st >>= 1) { if (t < st) red[t] += red[t + st]; __syncthreads(); }
    float ssum = red[0]; __syncthreads();
    float o[16];
#pragma unroll
    for (int d = 0; d < 16; ++d) o[d] = 0.f;
    for (int n = t; n < HW; n += 256) {
        float p = probs[n];
        const float* vp = v + ((size_t)b * HW + n) * 128 + h * 16;
#pragma unroll
        for (int d = 0; d < 16; ++d) o[d] += p * vp[d];
    }
    for (int d = 0; d < 16; ++d) {
        red[t] = o[d]; __syncthreads();
        for (int st = 128; st > 0; st >>= 1) { if (t < st) red[t] += red[t + st]; __syncthreads(); }
        if (t == 0) outp[b * 128 + h * 16 + d] = red[0] / ssum;
        __syncthreads();
    }
}

// keys = LN(keys + crow[b]); refresh f16 copies
__global__ void keys_update(float* __restrict__ keys, const float* __restrict__ crow,
                            const float* __restrict__ sc, const float* __restrict__ bi,
                            const float* __restrict__ pe, _Float16* __restrict__ kh,
                            _Float16* __restrict__ kpeh) {
    __shared__ float red[256];
    size_t row = blockIdx.x; int t = threadIdx.x;
    int b = (int)(row >> 12), n = (int)(row & 4095);
    size_t o = row * DD + t;
    float v = keys[o] + crow[b * DD + t];
    red[t] = v; __syncthreads();
    for (int st = 128; st > 0; st >>= 1) { if (t < st) red[t] += red[t + st]; __syncthreads(); }
    float m = red[0] / 256.f; __syncthreads();
    float d = v - m;
    red[t] = d * d; __syncthreads();
    for (int st = 128; st > 0; st >>= 1) { if (t < st) red[t] += red[t + st]; __syncthreads(); }
    float var = red[0] / 256.f;
    float y = d * rsqrtf(var + 1e-5f) * sc[t] + bi[t];
    keys[o] = y;
    kh[o] = (_Float16)y;
    kpeh[o] = (_Float16)(y + pe[(size_t)n * DD + t]);
}

__device__ __forceinline__ float gelu_exact(float x) {
    return 0.5f * x * (1.f + erff(x * 0.70710678f));
}

// convT1 output rows (B*4096 x 256, col = o*4 + dy*2 + dx) -> LN over o, gelu,
// scatter to up1h[b, pix(128x128), o] f16
__global__ void ln2d_gelu(const float* __restrict__ tmp1, const float* __restrict__ cb,
                          const float* __restrict__ sc, const float* __restrict__ bi,
                          _Float16* __restrict__ up1h) {
    int i = blockIdx.x * 256 + threadIdx.x;      // B*4096*4
    if (i >= BATCH * HW * 4) return;
    int j = i & 3, n = (i >> 2) & 4095, b = i >> 14;
    const float* base = tmp1 + ((size_t)b * HW + n) * 256 + j;
    float sum = 0.f, sq = 0.f;
    for (int o = 0; o < 64; ++o) { float x = base[4 * o] + cb[o]; sum += x; sq += x * x; }
    float m = sum / 64.f;
    float var = sq / 64.f - m * m;
    float inv = rsqrtf(var + 1e-6f);
    int h = n >> 6, w = n & 63, dy = j >> 1, dx = j & 1;
    int pix = (2 * h + dy) * 128 + 2 * w + dx;
    _Float16* op = up1h + ((size_t)b * 16384 + pix) * 64;
    for (int o = 0; o < 64; ++o) {
        float x = base[4 * o] + cb[o];
        float y = (x - m) * inv * sc[o] + bi[o];
        op[o] = (_Float16)gelu_exact(y);
    }
}

// convT2 rows (B*16384 x 128, col = o2*4 + j2) -> gelu -> dot with hyp -> heat
__global__ void heat_kernel(const float* __restrict__ tmp2, const float* __restrict__ cb,
                            const float* __restrict__ hyp, const float* __restrict__ masks,
                            float* __restrict__ heat) {
    int i = blockIdx.x * 256 + threadIdx.x;      // B*16384*4
    if (i >= BATCH * 16384 * 4) return;
    int j = i & 3, pix = (i >> 2) & 16383, b = i >> 16;
    const float* base = tmp2 + ((size_t)b * 16384 + pix) * 128 + j;
    float acc = 0.f;
    for (int o = 0; o < 32; ++o) {
        float x = base[4 * o] + cb[o];
        acc += hyp[b * 32 + o] * gelu_exact(x);
    }
    int y1 = pix >> 7, x1 = pix & 127;
    int y2 = 2 * y1 + (j >> 1), x2 = 2 * x1 + (j & 1);
    size_t hi = (size_t)b * 65536 + y2 * 256 + x2;
    heat[hi] = acc * masks[hi];
}

// 3x3 NMS + threshold + pack sort keys (value desc, index asc)
__global__ void nms_kernel(const float* __restrict__ heat, float* __restrict__ nmsout,
                           unsigned long long* __restrict__ keys) {
    int i = blockIdx.x * 256 + threadIdx.x;      // B*65536
    if (i >= BATCH * 65536) return;
    int b = i >> 16, p = i & 65535;
    int y = p >> 8, x = p & 255;
    const float* hb = heat + (size_t)b * 65536;
    float c = hb[p];
    float mx = -1e30f;
    for (int dy = -1; dy <= 1; ++dy) {
        int yy = y + dy; if (yy < 0 || yy > 255) continue;
        for (int dx = -1; dx <= 1; ++dx) {
            int xx = x + dx; if (xx < 0 || xx > 255) continue;
            mx = fmaxf(mx, hb[yy * 256 + xx]);
        }
    }
    float nv = (mx == c) ? c : 0.f;
    nmsout[(size_t)b * 65536 + p] = nv;
    float mv = (nv > THRV) ? nv : 0.f;           // mv >= 0 -> float bits monotonic
    unsigned int bits = __float_as_uint(mv);
    keys[(size_t)b * 65536 + p] =
        ((unsigned long long)bits << 32) | (unsigned int)(0xFFFFFFFFu - (unsigned)p);
}

// per-batch in-place bitonic sort (descending), one block owns 65536 keys
__global__ void bitonic_kernel(unsigned long long* __restrict__ keys) {
    unsigned long long* A = keys + (size_t)blockIdx.x * 65536;
    const int N = 65536;
    int t = threadIdx.x;
    for (int k = 2; k <= N; k <<= 1) {
        for (int j = k >> 1; j > 0; j >>= 1) {
            for (int i = t; i < N; i += 1024) {
                int ixj = i ^ j;
                if (ixj > i) {
                    unsigned long long a = A[i], b = A[ixj];
                    bool up = (i & k) == 0;
                    bool sw = up ? (a < b) : (a > b);   // descending overall
                    if (sw) { A[i] = b; A[ixj] = a; }
                }
            }
            __syncthreads();
        }
    }
}

__global__ void points_kernel(const unsigned long long* __restrict__ keys,
                              float* __restrict__ pts, float* __restrict__ scores) {
    int i = blockIdx.x * 256 + threadIdx.x;
    if (i >= BATCH * 1000) return;
    int b = i / 1000, r = i % 1000;
    unsigned long long key = keys[(size_t)b * 65536 + r];
    float v = __uint_as_float((unsigned)(key >> 32));
    unsigned idx = 0xFFFFFFFFu - (unsigned)key;
    float px = 0.f, py = 0.f, sc = 0.f;
    if (v > THRV) {
        px = (float)(idx & 255) * 4.f;
        py = (float)(idx >> 8) * 4.f;
        sc = v;
    }
    pts[(size_t)(b * 1000 + r) * 2]     = px;
    pts[(size_t)(b * 1000 + r) * 2 + 1] = py;
    scores[b * 1000 + r] = sc;
}

// ---------------------------------------------------------------------------
extern "C" void kernel_launch(void* const* d_in, const int* in_sizes, int n_in,
                              void* d_out, int out_size, void* d_ws, size_t ws_size,
                              hipStream_t stream) {
    auto P = [&](int idx) { return (const float*)d_in[idx]; };
    const float* img   = P(0);
    const float* masks = P(1);
    // param index helpers (p-dict insertion order, offset by 2 inputs)
    auto LW = [&](int layer, int lin) { return P(2 + 36 * layer + 2 * lin); };
    auto LB = [&](int layer, int lin) { return P(2 + 36 * layer + 2 * lin + 1); };
    auto LN_S = [&](int layer, int n) { return P(2 + 36 * layer + 28 + 2 * n); };
    auto LN_B = [&](int layer, int n) { return P(2 + 36 * layer + 28 + 2 * n + 1); };
    // linear index within layer: sa_q 0, sa_k 1, sa_v 2, sa_o 3, t2i_q 4, t2i_k 5,
    // t2i_v 6, t2i_o 7, mlp1 8, mlp2 9, i2t_q 10, i2t_k 11, i2t_v 12, i2t_o 13
    const float* fin_q_w = P(74); const float* fin_q_b = P(75);
    const float* fin_k_w = P(76); const float* fin_k_b = P(77);
    const float* fin_v_w = P(78); const float* fin_v_b = P(79);
    const float* fin_o_w = P(80); const float* fin_o_b = P(81);
    const float* fn_s = P(82);    const float* fn_b = P(83);
    const float* mask_token = P(84);
    const float* ct1_w = P(85);   const float* ct1_b = P(86);
    const float* ln2d_s = P(87);  const float* ln2d_b = P(88);
    const float* ct2_w = P(89);   const float* ct2_b = P(90);
    const float* hyp_w[3] = { P(91), P(93), P(95) };
    const float* hyp_b[3] = { P(92), P(94), P(96) };
    const float* pe_gauss = P(97);

    // -------- workspace layout --------
    char* ws = (char*)d_ws;
    size_t o_pe    = 0;
    size_t o_keys  = o_pe    + (size_t)4 * HW * DD;               // 4 MB
    size_t o_keysh = o_keys  + (size_t)4 * BATCH * HW * DD;       // +134 MB
    size_t o_kpeh  = o_keysh + (size_t)2 * BATCH * HW * DD;       // +67 MB
    size_t o_proj  = o_kpeh  + (size_t)2 * BATCH * HW * DD;       // +67 MB
    size_t o_tmp2  = o_proj  + (size_t)4 * BATCH * HW * DD;       // +134 MB (projk|projv, aliased by tmp1)
    size_t o_sort  = o_tmp2  + (size_t)4 * BATCH * 16384 * 128;   // +268 MB
    size_t o_wt    = o_sort  + (size_t)8 * BATCH * 65536;         // +16 MB
    size_t o_small = o_wt    + 2 * (size_t)2 * 256 * 256;         // two WT slots

    float*    pe      = (float*)(ws + o_pe);
    float*    keys    = (float*)(ws + o_keys);
    _Float16* keysh   = (_Float16*)(ws + o_keysh);
    _Float16* kpeh    = (_Float16*)(ws + o_kpeh);
    float*    projk   = (float*)(ws + o_proj);
    float*    projv   = (float*)(ws + o_proj + (size_t)4 * BATCH * HW * 128);
    float*    tmp1    = (float*)(ws + o_proj);                    // alias (decoder phase)
    _Float16* up1h    = (_Float16*)(ws + o_kpeh);                 // alias (decoder phase)
    float*    tmp2    = (float*)(ws + o_tmp2);
    unsigned long long* sortkeys = (unsigned long long*)(ws + o_sort);
    _Float16* wt_a    = (_Float16*)(ws + o_wt);
    _Float16* wt_b    = (_Float16*)(ws + o_wt + (size_t)2 * 256 * 256);
    float*    queries = (float*)(ws + o_small);
    float*    t1      = queries + BATCH * DD;
    float*    t2      = t1 + BATCH * DD;
    float*    q32     = t2 + BATCH * DD;
    float*    attnout = q32 + BATCH * 128;
    float*    hypbuf  = attnout + BATCH * 128;

    float* out_heat   = (float*)d_out;
    float* out_pts    = out_heat + (size_t)BATCH * 65536;
    float* out_scores = out_pts + (size_t)BATCH * 1000 * 2;
    float* out_nms    = out_scores + (size_t)BATCH * 1000;

    auto run_gemm = [&](const _Float16* A, const float* Wf, const float* bias,
                        float* C, int M, int N, int K, _Float16* wtbuf) {
        int tot = K * N;
        transpose_w<<<(tot + 255) / 256, 256, 0, stream>>>(Wf, wtbuf, K, N);
        dim3 g(N / 128, M / 64);                  // NT = 8 -> 128 cols per block-x
        if (K == 256)
            gemm_wmma_f16<8, 256><<<g, 128, 0, stream>>>(A, wtbuf, bias, C, M, N);
        else
            gemm_wmma_f16<8, 64><<<g, 128, 0, stream>>>(A, wtbuf, bias, C, M, N);
    };

    // -------- init --------
    pe_kernel<<<(HW * 128 + 255) / 256, 256, 0, stream>>>(pe, pe_gauss);
    {
        dim3 g(HW / 32, DD / 32, BATCH), blk(32, 8);
        keys_init<<<g, blk, 0, stream>>>(img, pe, keys, keysh, kpeh);
    }
    copy_queries<<<(BATCH * DD + 255) / 256, 256, 0, stream>>>(queries, mask_token);

    const int Mtok = BATCH * HW;   // 131072

    // -------- transformer layers --------
    for (int L = 0; L < 2; ++L) {
        // self-attention on 1 token: softmax==1 -> (q @ v_w + v_b) @ o_w + o_b
        lin_small<<<BATCH, 256, 0, stream>>>(queries, LW(L, 2), LB(L, 2), t1, 256, 256, nullptr, 0);
        lin_small<<<BATCH, 256, 0, stream>>>(t1, LW(L, 3), LB(L, 3), t2, 256, 256, nullptr, 0);
        res_ln<<<BATCH, 256, 0, stream>>>(queries, t2, LN_S(L, 0), LN_B(L, 0), L == 0 ? 1 : 0);

        // t2i cross attention
        lin_small<<<BATCH, 256, 0, stream>>>(queries, LW(L, 4), LB(L, 4), q32, 256, 128, mask_token, 0);
        run_gemm(kpeh,  LW(L, 5), LB(L, 5), projk, Mtok, 128, 256, wt_a);   // k = (keys+pe) @ k_w
        run_gemm(keysh, LW(L, 6), LB(L, 6), projv, Mtok, 128, 256, wt_b);   // v = keys @ v_w
        attn_t2i<<<BATCH * 8, 256, 0, stream>>>(q32, projk, projv, attnout);
        lin_small<<<BATCH, 256, 0, stream>>>(attnout, LW(L, 7), LB(L, 7), t1, 128, 256, nullptr, 0);
        res_ln<<<BATCH, 256, 0, stream>>>(queries, t1, LN_S(L, 1), LN_B(L, 1), 0);

        // MLP + LN
        mlp_kernel<<<BATCH, 256, 0, stream>>>(queries, LW(L, 8), LB(L, 8),
                                              LW(L, 9), LB(L, 9), LN_S(L, 2), LN_B(L, 2));

        // i2t: single key -> softmax==1 -> constant row per batch
        lin_small<<<BATCH, 256, 0, stream>>>(queries, LW(L, 12), LB(L, 12), t1, 256, 128, nullptr, 0);
        lin_small<<<BATCH, 256, 0, stream>>>(t1, LW(L, 13), LB(L, 13), t2, 128, 256, nullptr, 0);
        keys_update<<<BATCH * HW, 256, 0, stream>>>(keys, t2, LN_S(L, 3), LN_B(L, 3),
                                                    pe, keysh, kpeh);
    }

    // -------- final token-to-image attention --------
    lin_small<<<BATCH, 256, 0, stream>>>(queries, fin_q_w, fin_q_b, q32, 256, 128, mask_token, 0);
    run_gemm(kpeh,  fin_k_w, fin_k_b, projk, Mtok, 128, 256, wt_a);
    run_gemm(keysh, fin_v_w, fin_v_b, projv, Mtok, 128, 256, wt_b);
    attn_t2i<<<BATCH * 8, 256, 0, stream>>>(q32, projk, projv, attnout);
    lin_small<<<BATCH, 256, 0, stream>>>(attnout, fin_o_w, fin_o_b, t1, 128, 256, nullptr, 0);
    res_ln<<<BATCH, 256, 0, stream>>>(queries, t1, fn_s, fn_b, 0);

    // -------- hypernetwork on final mask token --------
    lin_small<<<BATCH, 256, 0, stream>>>(queries, hyp_w[0], hyp_b[0], t1, 256, 256, nullptr, 1);
    lin_small<<<BATCH, 256, 0, stream>>>(t1, hyp_w[1], hyp_b[1], t2, 256, 256, nullptr, 1);
    lin_small<<<BATCH, 256, 0, stream>>>(t2, hyp_w[2], hyp_b[2], hypbuf, 256, 32, nullptr, 0);

    // -------- upsampling decoder as GEMMs --------
    // convT1: A = keys (B*4096 x 256), W view (256 x 256), bias folded into LN stage
    run_gemm(keysh, ct1_w, nullptr, tmp1, Mtok, 256, 256, wt_a);
    ln2d_gelu<<<(BATCH * HW * 4 + 255) / 256, 256, 0, stream>>>(tmp1, ct1_b, ln2d_s, ln2d_b, up1h);
    // convT2: A = up1h (B*16384 x 64), W view (64 x 128)
    run_gemm(up1h, ct2_w, nullptr, tmp2, BATCH * 16384, 128, 64, wt_b);
    heat_kernel<<<(BATCH * 16384 * 4 + 255) / 256, 256, 0, stream>>>(tmp2, ct2_b, hypbuf,
                                                                     masks, out_heat);

    // -------- NMS, sort, points --------
    nms_kernel<<<(BATCH * 65536 + 255) / 256, 256, 0, stream>>>(out_heat, out_nms, sortkeys);
    bitonic_kernel<<<BATCH, 1024, 0, stream>>>(sortkeys);
    points_kernel<<<(BATCH * 1000 + 255) / 256, 256, 0, stream>>>(sortkeys, out_pts, out_scores);

    (void)in_sizes; (void)n_in; (void)out_size; (void)ws_size;
}
